// CrossAttention_47004122087816
// MI455X (gfx1250) — matrix operand order repaired and checked by reading, not compiled, not run on hardware
//
#include <hip/hip_runtime.h>
#include <hip/hip_bf16.h>

// ---------------------------------------------------------------------------
// CDNA5 / gfx1250 cross-attention (colsum-fused) implementation, bf16 WMMA
// + TDM (tensor_load_to_lds) staging of the q tile in the attention kernel.
// Dual accumulator chains per wave hide WMMA->WMMA RAW hazard NOPs.
// ---------------------------------------------------------------------------

typedef unsigned short u16t;
typedef __attribute__((ext_vector_type(16))) __bf16 v16bf;
typedef __attribute__((ext_vector_type(8)))  float  v8f;
typedef __attribute__((ext_vector_type(4)))  unsigned int uvec4;

// TDM descriptor group types (this toolchain: 6-arg clang-23 form)
typedef __attribute__((ext_vector_type(4))) unsigned int tdm_u32x4;
typedef __attribute__((ext_vector_type(8))) int          tdm_i32x8;
typedef __attribute__((ext_vector_type(4))) int          tdm_i32x4;

union FragBF { v16bf bf; uvec4 u[2]; };

// fp32 -> bf16 round-to-nearest-even
__device__ inline u16t f2bf(float f) {
  unsigned int u = __float_as_uint(f);
  unsigned int r = u + 0x7FFFu + ((u >> 16) & 1u);
  return (u16t)(r >> 16);
}

// A-matrix 16x32 bf16 fragment (ISA 7.12.2): lane m<16 -> row m, K = {0..7, 16..23};
// lane m+16 -> row m, K = {8..15, 24..31}. Two 16B vector loads.
__device__ inline v16bf load_frag_A(const u16t* base, int ld, int kc, int lane) {
  const int m  = lane & 15;
  const int hi = lane >> 4;
  const u16t* p = base + (size_t)m * ld + kc + hi * 8;
  FragBF f;
  f.u[0] = *reinterpret_cast<const uvec4*>(p);
  f.u[1] = *reinterpret_cast<const uvec4*>(p + 16);
  return f.bf;
}

// B-matrix 32x16 bf16 fragment, B supplied transposed (row n = column n of B):
// lane n<16 -> col n, K = {0..7, 16..23}; lane n+16 -> col n, K = {8..15, 24..31}.
__device__ inline v16bf load_frag_B(const u16t* base, int ld, int kc, int lane) {
  const int n  = lane & 15;
  const int hi = lane >> 4;
  const u16t* p = base + (size_t)n * ld + kc + hi * 8;
  FragBF f;
  f.u[0] = *reinterpret_cast<const uvec4*>(p);
  f.u[1] = *reinterpret_cast<const uvec4*>(p + 16);
  return f.bf;
}

__device__ inline v8f wmma_bf16(v16bf a, v16bf b, v8f c) {
  return __builtin_amdgcn_wmma_f32_16x16x32_bf16(false, a, false, b,
                                                 (short)0, c, false, false);
}

// ---------------------------------------------------------------------------
// TDM: DMA a 2D tile (rows x width_units, data_size = 8B units) from global
// memory into LDS. D# layout per CDNA5 ISA sec 8.3/8.4. Issued once per wave.
// ---------------------------------------------------------------------------
__device__ inline void tdm_load_2d_to_lds(unsigned lds_byte_off,
                                          unsigned long long gaddr,
                                          unsigned width_units,   // dim0 (8B units)
                                          unsigned rows,          // dim1
                                          unsigned row_stride_units) {
  tdm_u32x4 g0;
  g0[0] = 1u;                                        // count=1, user descriptor
  g0[1] = lds_byte_off;                              // lds_addr[31:0]
  g0[2] = (unsigned)(gaddr & 0xFFFFFFFFull);         // global_addr[31:0]
  g0[3] = (unsigned)((gaddr >> 32) & 0x1FFFFFFull)   // global_addr[56:32]
          | (2u << 30);                              // type=2 ("image")
  tdm_i32x8 g1;
  g1[0] = (int)(3u << 16);                           // wg_mask=0, data_size=3 (8B)
  g1[1] = (int)((width_units & 0xFFFFu) << 16);      // tensor_dim0[15:0]
  g1[2] = (int)(((width_units >> 16) & 0xFFFFu)      // tensor_dim0[31:16]
          | ((rows & 0xFFFFu) << 16));               // tensor_dim1[15:0]
  g1[3] = (int)(((rows >> 16) & 0xFFFFu)             // tensor_dim1[31:16]
          | ((width_units & 0xFFFFu) << 16));        // tile_dim0
  g1[4] = (int)(rows & 0xFFFFu);                     // tile_dim1 (tile_dim2=0)
  g1[5] = (int)row_stride_units;                     // tensor_dim0_stride[31:0]
  g1[6] = 0;                                         // stride hi / dim1_stride lo
  g1[7] = 0;
  tdm_i32x4 z4 = {0, 0, 0, 0};                       // groups 2/3: NULL (2D tensor)
  tdm_i32x8 z8 = {0, 0, 0, 0, 0, 0, 0, 0};
  __builtin_amdgcn_tensor_load_to_lds(g0, g1, z4, z4, z8, 0);
}

// ---------------------------------------------------------------------------
// Small conversion / init kernels
// ---------------------------------------------------------------------------

__global__ void cvt_bf16_kernel(const float* __restrict__ in, u16t* __restrict__ out, size_t n) {
  size_t i = (size_t)blockIdx.x * 256 + threadIdx.x;
  if (i < n) out[i] = f2bf(in[i]);
}

// W is K x N row-major; WT is N x K row-major bf16.
__global__ void transpose_cvt_kernel(const float* __restrict__ W, u16t* __restrict__ WT,
                                     int K, int N) {
  int i = blockIdx.x * 256 + threadIdx.x;
  if (i >= K * N) return;
  int n = i / K, k = i % K;
  WT[i] = f2bf(W[(size_t)k * N + n]);
}

__global__ void zero_kernel(float* __restrict__ p, int n) {
  int i = blockIdx.x * 256 + threadIdx.x;
  if (i < n) p[i] = 0.0f;
}

// ---------------------------------------------------------------------------
// LayerNorm (one wave32 per 1024-elem row) -> bf16
// ---------------------------------------------------------------------------

__global__ void __launch_bounds__(256)
ln_bf16_kernel(const float* __restrict__ x, const float* __restrict__ w,
               const float* __restrict__ b, u16t* __restrict__ xn, int nrows) {
  const int row  = blockIdx.x * 8 + (threadIdx.x >> 5);
  const int lane = threadIdx.x & 31;
  if (row >= nrows) return;
  const float4* xr = reinterpret_cast<const float4*>(x + (size_t)row * 1024);
  float4 v[8];
  float s = 0.0f, s2 = 0.0f;
#pragma unroll
  for (int i = 0; i < 8; ++i) {
    v[i] = xr[lane + i * 32];
    s  += v[i].x + v[i].y + v[i].z + v[i].w;
    s2 += v[i].x * v[i].x + v[i].y * v[i].y + v[i].z * v[i].z + v[i].w * v[i].w;
  }
#pragma unroll
  for (int m = 1; m < 32; m <<= 1) {
    s  += __shfl_xor(s,  m, 32);
    s2 += __shfl_xor(s2, m, 32);
  }
  const float mu   = s * (1.0f / 1024.0f);
  const float var  = s2 * (1.0f / 1024.0f) - mu * mu;
  const float rstd = rsqrtf(var + 1e-5f);
  const float4* wv = reinterpret_cast<const float4*>(w);
  const float4* bv = reinterpret_cast<const float4*>(b);
  ushort4* outr = reinterpret_cast<ushort4*>(xn + (size_t)row * 1024);
#pragma unroll
  for (int i = 0; i < 8; ++i) {
    const float4 ww = wv[lane + i * 32];
    const float4 bb = bv[lane + i * 32];
    ushort4 o;
    o.x = f2bf((v[i].x - mu) * rstd * ww.x + bb.x);
    o.y = f2bf((v[i].y - mu) * rstd * ww.y + bb.y);
    o.z = f2bf((v[i].z - mu) * rstd * ww.z + bb.z);
    o.w = f2bf((v[i].w - mu) * rstd * ww.w + bb.w);
    outr[lane + i * 32] = o;
  }
}

// ---------------------------------------------------------------------------
// Generic wave-tile bf16 GEMM: C[M,N] = alpha * A[M,K] @ BT[N,K]^T
// One wave computes one 16x32 C tile (two 16x16 WMMA accumulator chains to
// hide WMMA->WMMA RAW hazards and reuse the A fragment). N % 32 == 0.
// Optional per-row scale (colsum) and f32/bf16 output.
// ---------------------------------------------------------------------------

template <bool ROWSCALE, bool OUT_BF16>
__global__ void __launch_bounds__(256)
gemm_bf16_kernel(const u16t* __restrict__ A, int lda,
                 const u16t* __restrict__ BT, int ldb,
                 void* __restrict__ Cv, int ldc,
                 const float* __restrict__ rowscale, float alpha,
                 int M, int N, int K) {
  const int lane  = threadIdx.x & 31;
  const int tile  = blockIdx.x * 8 + (threadIdx.x >> 5);
  const int tilesN2 = N >> 5;                 // pairs of 16-col tiles
  const int mt  = tile / tilesN2;
  const int nt2 = tile - mt * tilesN2;
  if (mt * 16 >= M) return;
  const u16t* Ab  = A + (size_t)mt * 16 * lda;
  const u16t* Bb0 = BT + (size_t)(nt2 * 2 + 0) * 16 * ldb;
  const u16t* Bb1 = BT + (size_t)(nt2 * 2 + 1) * 16 * ldb;
  v8f acc0 = {}, acc1 = {};
#pragma unroll 4
  for (int kc = 0; kc < K; kc += 32) {
    v16bf af = load_frag_A(Ab, lda, kc, lane);
    v16bf b0 = load_frag_B(Bb0, ldb, kc, lane);
    v16bf b1 = load_frag_B(Bb1, ldb, kc, lane);
    acc0 = wmma_bf16(af, b0, acc0);
    acc1 = wmma_bf16(af, b1, acc1);
  }
  // C layout: lane n<16 -> col n / rows 0..7; lane n+16 -> col n / rows 8..15
  const int col0 = nt2 * 32 + (lane & 15);
  const int rb   = mt * 16 + ((lane >> 4) << 3);
#pragma unroll
  for (int r = 0; r < 8; ++r) {
    const int row = rb + r;
    float v0 = acc0[r] * alpha;
    float v1 = acc1[r] * alpha;
    if (ROWSCALE) { const float rs = rowscale[row]; v0 *= rs; v1 *= rs; }
    if (OUT_BF16) {
      u16t* C = reinterpret_cast<u16t*>(Cv);
      C[(size_t)row * ldc + col0]      = f2bf(v0);
      C[(size_t)row * ldc + col0 + 16] = f2bf(v1);
    } else {
      float* C = reinterpret_cast<float*>(Cv);
      C[(size_t)row * ldc + col0]      = v0;
      C[(size_t)row * ldc + col0 + 16] = v1;
    }
  }
}

// ---------------------------------------------------------------------------
// Attention column-sum kernel. One 64-thread WG (2 waves) owns 32 q-rows
// staged in LDS via TDM. Pass 1: online per-row (max,sum) over all 2048 cols.
// Pass 2: recompute sim tiles, accumulate softmax column sums via f32 atomics.
// Two j-tiles per iteration -> two independent WMMA chains, shared A fragment.
// ---------------------------------------------------------------------------

__device__ inline void sim_tile2(const u16t* arow, const u16t* kbase, int jt, int lane,
                                 v8f& acc0, v8f& acc1) {
  acc0 = (v8f){};
  acc1 = (v8f){};
  const u16t* kb0 = kbase + (size_t)(jt + 0) * (16 * 1024);
  const u16t* kb1 = kbase + (size_t)(jt + 1) * (16 * 1024);
#pragma unroll 4
  for (int kc = 0; kc < 512; kc += 32) {
    v16bf af = load_frag_A(arow, 512, kc, lane);   // LDS (TDM-staged)
    v16bf b0 = load_frag_B(kb0, 1024, kc, lane);   // global (L2-resident k)
    v16bf b1 = load_frag_B(kb1, 1024, kc, lane);
    acc0 = wmma_bf16(af, b0, acc0);
    acc1 = wmma_bf16(af, b1, acc1);
  }
}

__global__ void __launch_bounds__(64)
attn_colsum_kernel(const u16t* __restrict__ q, const u16t* __restrict__ kv,
                   float* __restrict__ colsum) {
  __shared__ u16t qs[32 * 512];  // 32 KB
  const int b    = blockIdx.y;
  const int m0   = blockIdx.x * 32;
  const int lane = threadIdx.x & 31;
  const int wave = threadIdx.x >> 5;  // 0..1, each owns a 16-row strip

  // TDM stage of the 32x512 bf16 q tile (32 rows x 1024B = 128 units of 8B).
  // Wave 0 issues the DMA (one op per wave; TDM ignores EXEC) and waits on
  // TENSORcnt before the barrier releases wave 1's ds_load reads.
  if (wave == 0) {
    const unsigned long long gq =
        (unsigned long long)(uintptr_t)(q + ((size_t)b * 2048 + m0) * 512);
    // low 32 bits of a generic LDS pointer = LDS byte offset (ISA 10.2)
    tdm_load_2d_to_lds((unsigned)(uintptr_t)qs, gq, 128u, 32u, 128u);
    __builtin_amdgcn_s_wait_tensorcnt(0);
  }
  __syncthreads();
  // Make the DMA's LDS write visible to the optimizer: qs escapes + memory
  // clobber, so the A-fragment ds_loads below cannot be treated as undef.
  asm volatile("" : : "r"(&qs[0]) : "memory");

  const u16t* kbase = kv + (size_t)b * 2048 * 1024;  // k = cols 0..511 of kv, ld=1024
  const u16t* arow  = qs + wave * 16 * 512;

  float rmax[8], rsum[8];
#pragma unroll
  for (int r = 0; r < 8; ++r) { rmax[r] = -3.0e38f; rsum[r] = 0.0f; }

  // ---- pass 1: per-row online max & exp-sum ----
  for (int jt = 0; jt < 128; jt += 2) {
    if (jt + 2 < 128)
      __builtin_prefetch(kbase + (size_t)(jt + 2) * (16 * 1024), 0, 3);
    v8f acc0, acc1;
    sim_tile2(arow, kbase, jt, lane, acc0, acc1);
#pragma unroll
    for (int t = 0; t < 2; ++t) {
      const v8f& acc = t ? acc1 : acc0;
#pragma unroll
      for (int r = 0; r < 8; ++r) {
        float v = acc[r];
        // row-wise reduce across the 16 columns (stay within 16-lane half)
        float tm = v;
        tm = fmaxf(tm, __shfl_xor(tm, 1, 32));
        tm = fmaxf(tm, __shfl_xor(tm, 2, 32));
        tm = fmaxf(tm, __shfl_xor(tm, 4, 32));
        tm = fmaxf(tm, __shfl_xor(tm, 8, 32));
        const float nm = fmaxf(rmax[r], tm);
        float e = __expf(v - nm);
        e += __shfl_xor(e, 1, 32);
        e += __shfl_xor(e, 2, 32);
        e += __shfl_xor(e, 4, 32);
        e += __shfl_xor(e, 8, 32);
        rsum[r] = rsum[r] * __expf(rmax[r] - nm) + e;
        rmax[r] = nm;
      }
    }
  }
  float rinv[8];
#pragma unroll
  for (int r = 0; r < 8; ++r) rinv[r] = 1.0f / rsum[r];

  // ---- pass 2: recompute, normalize, column-sum, atomic accumulate ----
  for (int jt = 0; jt < 128; jt += 2) {
    if (jt + 2 < 128)
      __builtin_prefetch(kbase + (size_t)(jt + 2) * (16 * 1024), 0, 3);
    v8f acc0, acc1;
    sim_tile2(arow, kbase, jt, lane, acc0, acc1);
    float cs0 = 0.0f, cs1 = 0.0f;
#pragma unroll
    for (int r = 0; r < 8; ++r) {
      cs0 += __expf(acc0[r] - rmax[r]) * rinv[r];
      cs1 += __expf(acc1[r] - rmax[r]) * rinv[r];
    }
    cs0 += __shfl_xor(cs0, 16, 32);  // fold rows 8..15 half onto rows 0..7 half
    cs1 += __shfl_xor(cs1, 16, 32);
    if (lane < 16) {
      atomicAdd(&colsum[(size_t)b * 2048 + (jt + 0) * 16 + lane], cs0);
      atomicAdd(&colsum[(size_t)b * 2048 + (jt + 1) * 16 + lane], cs1);
    }
  }
}

// ---------------------------------------------------------------------------
// Host-side orchestration
// ---------------------------------------------------------------------------

extern "C" void kernel_launch(void* const* d_in, const int* in_sizes, int n_in,
                              void* d_out, int out_size, void* d_ws, size_t ws_size,
                              hipStream_t stream) {
  (void)in_sizes; (void)n_in; (void)out_size; (void)ws_size;
  const float* x     = (const float*)d_in[0];
  const float* media = (const float*)d_in[1];
  const float* ln_w  = (const float*)d_in[2];
  const float* ln_b  = (const float*)d_in[3];
  const float* Wq    = (const float*)d_in[4];
  const float* Wkv   = (const float*)d_in[5];
  const float* Wout  = (const float*)d_in[6];
  float* out = (float*)d_out;

  const int B = 8, N = 2048, D = 1024, I = 512;
  const size_t rows = (size_t)B * N;  // 16384

  char* ws = (char*)d_ws;
  u16t* WqT   = (u16t*)ws; ws += (size_t)I * D * 2;        // 512 x 1024 bf16
  u16t* WkvT  = (u16t*)ws; ws += (size_t)(2 * I) * D * 2;  // 1024 x 1024 bf16
  u16t* WoutT = (u16t*)ws; ws += (size_t)D * I * 2;        // 1024 x 512 bf16
  u16t* xnb   = (u16t*)ws; ws += rows * D * 2;             // LN(x) bf16
  u16t* medb  = (u16t*)ws; ws += rows * D * 2;             // media bf16
  u16t* qb    = (u16t*)ws; ws += rows * I * 2;             // q bf16 (pre-scaled)
  u16t* kvb   = (u16t*)ws; ws += rows * (2 * I) * 2;       // kv bf16 (k | v)
  float* colsum = (float*)ws;                              // 16384 f32

  // Weight transposes + bf16 converts
  transpose_cvt_kernel<<<(I * D + 255) / 256, 256, 0, stream>>>(Wq, WqT, D, I);
  transpose_cvt_kernel<<<(2 * I * D + 255) / 256, 256, 0, stream>>>(Wkv, WkvT, D, 2 * I);
  transpose_cvt_kernel<<<(D * I + 255) / 256, 256, 0, stream>>>(Wout, WoutT, I, D);

  // media -> bf16
  cvt_bf16_kernel<<<(unsigned)((rows * D + 255) / 256), 256, 0, stream>>>(media, medb, rows * D);

  // LayerNorm(x) -> bf16
  ln_bf16_kernel<<<(unsigned)(rows / 8), 256, 0, stream>>>(x, ln_w, ln_b, xnb, (int)rows);

  // q = LN(x) @ Wq * SCALE   (M=16384, N=512, K=1024)
  gemm_bf16_kernel<false, true>
      <<<(unsigned)((rows / 16) * (I / 32) / 8), 256, 0, stream>>>(
          xnb, D, WqT, D, qb, I, nullptr, 0.125f, (int)rows, I, D);

  // kv = media @ Wkv         (M=16384, N=1024, K=1024)
  gemm_bf16_kernel<false, true>
      <<<(unsigned)((rows / 16) * ((2 * I) / 32) / 8), 256, 0, stream>>>(
          medb, D, WkvT, D, kvb, 2 * I, nullptr, 1.0f, (int)rows, 2 * I, D);

  // colsum = 0
  zero_kernel<<<(unsigned)((rows + 255) / 256), 256, 0, stream>>>(colsum, (int)rows);

  // attention softmax column sums
  dim3 ag(N / 32, B);
  attn_colsum_kernel<<<ag, 64, 0, stream>>>(qb, kvb, colsum);

  // y = diag(colsum) * (v @ Wout)   (A = v = kv cols 512.., lda=1024; M=16384, N=1024, K=512)
  gemm_bf16_kernel<true, false>
      <<<(unsigned)((rows / 16) * (D / 32) / 8), 256, 0, stream>>>(
          kvb + I, 2 * I, WoutT, I, out, D, colsum, 1.0f, (int)rows, D, I);
}
